// SNNBlock_42133629174355
// MI455X (gfx1250) — matrix operand order repaired and checked by reading, not compiled, not run on hardware
//
#include <hip/hip_runtime.h>

// SNN leaky-integrate-and-fire block, fused GEMM + scan for gfx1250 (MI455X).
//   cur = x @ W^T + b            (f16 WMMA, f32 accumulate)
//   mem = beta*mem + cur_t - (mem_prev > thr)*thr    [reset-by-subtract]
//   spk = (mem > thr)
// Pipeline:
//   kernel 1: x,W f32 -> f16 in d_ws (streamed, NT loads)
//   kernel 2: per (16-batch x 16-h) tile: W held in VGPRs as 16 B-fragments,
//             membrane state held in the WMMA C/D fragment for all 200 steps,
//             x tiles double-buffered in LDS via GLOBAL_LOAD_ASYNC_TO_LDS_B128
//             (ASYNCcnt), outputs stored non-temporally.

typedef __attribute__((ext_vector_type(16))) _Float16 v16h;
typedef __attribute__((ext_vector_type(8)))  _Float16 v8h;
typedef __attribute__((ext_vector_type(4)))  _Float16 v4h;
typedef __attribute__((ext_vector_type(8)))  float    v8f;
typedef __attribute__((ext_vector_type(4)))  float    v4f;

#define T_STEPS 200
#define BATCH   256
#define D_IN    512
#define D_H     512
#define BETA    0.9f
#define THR     1.0f

#define NX  ((size_t)T_STEPS * BATCH * D_IN)   // x elements
#define NW  ((size_t)D_H * D_IN)               // W elements

// ---------------------------------------------------------------------------
// Kernel 1: convert x and W to f16 in workspace. Exact grid, vectorized x4.
// ---------------------------------------------------------------------------
__global__ __launch_bounds__(256) void cvt_f16_kernel(
    const float* __restrict__ x, const float* __restrict__ W,
    _Float16* __restrict__ xh, _Float16* __restrict__ wh)
{
    const size_t i4  = (size_t)blockIdx.x * 256 + threadIdx.x; // float4 index
    const size_t nx4 = NX / 4;

    v4f v;
    _Float16* dst;
    if (i4 < nx4) {
        v   = __builtin_nontemporal_load(&((const v4f*)x)[i4]); // f32 x read once
        dst = xh + i4 * 4;
    } else {
        const size_t j = i4 - nx4;
        v   = ((const v4f*)W)[j];
        dst = wh + j * 4;
    }
    v4h h;
    h[0] = (_Float16)v[0];
    h[1] = (_Float16)v[1];
    h[2] = (_Float16)v[2];
    h[3] = (_Float16)v[3];
    *(v4h*)dst = h;
}

// ---------------------------------------------------------------------------
// Async copy helpers (CDNA5 async global->LDS path, tracked on ASYNCcnt).
// ---------------------------------------------------------------------------
__device__ __forceinline__ void async_copy_b128(unsigned lds_addr,
                                                const _Float16* gaddr) {
    asm volatile("global_load_async_to_lds_b128 %0, %1, off"
                 :: "v"(lds_addr), "v"(gaddr) : "memory");
}
__device__ __forceinline__ void wait_async0() {
    asm volatile("s_wait_asynccnt 0x0" ::: "memory");
}

// Stage one contiguous 16x512 f16 tile (16 KB) into LDS: 1024 16B chunks,
// 8 per thread, issued asynchronously (no VGPR data, no waits here).
__device__ __forceinline__ void stage_tile_async(const _Float16* __restrict__ src,
                                                 unsigned lds_base, int tid) {
    #pragma unroll
    for (int it = 0; it < 8; ++it) {
        const int idx = it * 128 + tid;            // 16-byte chunk index
        async_copy_b128(lds_base + (unsigned)idx * 16, src + (size_t)idx * 8);
    }
}

// ---------------------------------------------------------------------------
// Kernel 2: fused GEMM + LIF scan, f16 inputs from workspace.
// ---------------------------------------------------------------------------
__global__ __launch_bounds__(128) void snn_fused_async_kernel(
    const _Float16* __restrict__ xh,   // [T, B, D_IN] f16
    const _Float16* __restrict__ wh,   // [D_H, D_IN] f16
    const float*    __restrict__ bias, // [D_H] f32
    float* __restrict__ spk_out,       // [T, B, D_H]
    float* __restrict__ mem_out)       // [T, B, D_H]
{
    __shared__ __align__(16) _Float16 sx[2][16 * D_IN]; // double buffer, 32 KB

    const int tid  = threadIdx.x;
    const int lane = tid & 31;
    const int wave = tid >> 5;

    const int b0 = blockIdx.y * 16;             // batch-tile base row
    const int h0 = blockIdx.x * 64 + wave * 16; // this wave's h-tile base col

    const int n  = lane & 15;        // C column (h) / A row (m) for this lane
    const int hi = (lane >> 4) & 1;  // lane half: selects K sub-range

    const unsigned lds_base0 = (unsigned)(uintptr_t)&sx[0][0]; // low 32 = LDS off
    const unsigned lds_base1 = (unsigned)(uintptr_t)&sx[1][0];

    // ---- W fragments (B-matrix, 16 x 512) persistent in VGPRs, loaded as f16.
    // Contiguous k order used here (32x16): lane<16 -> N=lane,  K=kc*32+0..15;
    //                                lane>=16 -> N=lane-16, K=kc*32+16..31.
    v16h wfrag[16];
    {
        const _Float16* wp = wh + (size_t)(h0 + n) * D_IN + hi * 16;
        #pragma unroll
        for (int kc = 0; kc < 16; ++kc) {
            union { v16h v; v8h h[2]; } w;
            w.h[0] = *(const v8h*)(wp + kc * 32);
            w.h[1] = *(const v8h*)(wp + kc * 32 + 8);
            wfrag[kc] = w.v;
        }
    }

    const float bf = bias[h0 + n];   // bias for this lane's column
    v8f mem = {};                    // membrane state fragment, zero-init

    const size_t out_tstride = (size_t)BATCH * D_H;

    // Prologue: stage tile for t=0.
    stage_tile_async(xh + ((size_t)0 * BATCH + b0) * D_IN, lds_base0, tid);
    wait_async0();
    __syncthreads();

    for (int t = 0; t < T_STEPS; ++t) {
        const unsigned base_cur = (t & 1) ? lds_base1 : lds_base0;
        const _Float16* scur    = (t & 1) ? &sx[1][0] : &sx[0][0];
        const unsigned base_nxt = (t & 1) ? lds_base0 : lds_base1;
        (void)base_cur;

        // Issue async copy of the NEXT x tile while computing this one.
        if (t + 1 < T_STEPS)
            stage_tile_async(xh + ((size_t)(t + 1) * BATCH + b0) * D_IN,
                             base_nxt, tid);

        // Fold the leaky-update prologue into the accumulator:
        // C0 = beta*mem - (mem_prev > thr)*thr + bias.
        v8f c;
        #pragma unroll
        for (int r = 0; r < 8; ++r) {
            const float m  = mem[r];
            const float rs = (m > THR) ? THR : 0.0f;
            c[r] = BETA * m - rs + bf;
        }

        // 16 chained v_wmma_f32_16x16x32_f16 over K=512.
        // Same contiguous k order for A (16x32): lane<16 -> M=lane,  K=0..15;
        //                                lane>=16 -> M=lane-16, K=16..31
        #pragma unroll
        for (int kc = 0; kc < 16; ++kc) {
            const int base = n * D_IN + kc * 32 + hi * 16;
            union { v16h v; v8h h[2]; } a;
            a.h[0] = *(const v8h*)(scur + base);
            a.h[1] = *(const v8h*)(scur + base + 8);
            c = __builtin_amdgcn_wmma_f32_16x16x32_f16(
                    false, a.v, false, wfrag[kc], (short)0, c, false, false);
        }
        mem = c;

        // Fire + store from the C/D fragment layout (lane l, VGPR r ->
        // row m = r + 8*(l>=16), col = l&15). Non-temporal: outputs are
        // write-once streams (210 MB) and must not evict x_h/W from L2.
        float* sp = spk_out + (size_t)t * out_tstride;
        float* mp = mem_out + (size_t)t * out_tstride;
        #pragma unroll
        for (int r = 0; r < 8; ++r) {
            const int    mrow = r + hi * 8;
            const size_t off  = (size_t)(b0 + mrow) * D_H + (h0 + n);
            const float  mv   = mem[r];
            __builtin_nontemporal_store((mv > THR) ? 1.0f : 0.0f, &sp[off]);
            __builtin_nontemporal_store(mv, &mp[off]);
        }

        wait_async0();     // own async copies for t+1 done...
        __syncthreads();   // ...and everyone's, before consuming next buffer
    }
}

// ---------------------------------------------------------------------------
// Fallback (no workspace needed): inline f32->f16 staging through LDS.
// ---------------------------------------------------------------------------
__global__ __launch_bounds__(128) void snn_fused_f32_kernel(
    const float* __restrict__ x, const float* __restrict__ W,
    const float* __restrict__ bias,
    float* __restrict__ spk_out, float* __restrict__ mem_out)
{
    __shared__ __align__(16) _Float16 sx[16 * D_IN];

    const int tid  = threadIdx.x;
    const int lane = tid & 31;
    const int wave = tid >> 5;
    const int b0 = blockIdx.y * 16;
    const int h0 = blockIdx.x * 64 + wave * 16;
    const int n  = lane & 15;
    const int hi = (lane >> 4) & 1;

    v16h wfrag[16];
    {
        const float* wrow = W + (size_t)(h0 + n) * D_IN + hi * 16;
        #pragma unroll
        for (int kc = 0; kc < 16; ++kc) {
            const float* p = wrow + kc * 32;
            v16h wf;
            #pragma unroll
            for (int i = 0; i < 16; ++i) wf[i] = (_Float16)p[i];
            wfrag[kc] = wf;
        }
    }

    const float bf = bias[h0 + n];
    v8f mem = {};
    const size_t out_tstride = (size_t)BATCH * D_H;

    for (int t = 0; t < T_STEPS; ++t) {
        __syncthreads();
        {
            const float* xt = x + ((size_t)t * BATCH + b0) * D_IN;
            #pragma unroll
            for (int it = 0; it < 16; ++it) {
                const int idx4 = it * 128 + tid;
                const v4f v = ((const v4f*)xt)[idx4];
                v4h h;
                h[0] = (_Float16)v[0]; h[1] = (_Float16)v[1];
                h[2] = (_Float16)v[2]; h[3] = (_Float16)v[3];
                ((v4h*)sx)[idx4] = h;
            }
        }
        __syncthreads();

        v8f c;
        #pragma unroll
        for (int r = 0; r < 8; ++r) {
            const float m  = mem[r];
            const float rs = (m > THR) ? THR : 0.0f;
            c[r] = BETA * m - rs + bf;
        }
        #pragma unroll
        for (int kc = 0; kc < 16; ++kc) {
            const int base = n * D_IN + kc * 32 + hi * 16;
            union { v16h v; v8h h[2]; } a;
            a.h[0] = *(const v8h*)&sx[base];
            a.h[1] = *(const v8h*)&sx[base + 8];
            c = __builtin_amdgcn_wmma_f32_16x16x32_f16(
                    false, a.v, false, wfrag[kc], (short)0, c, false, false);
        }
        mem = c;

        float* sp = spk_out + (size_t)t * out_tstride;
        float* mp = mem_out + (size_t)t * out_tstride;
        #pragma unroll
        for (int r = 0; r < 8; ++r) {
            const int    mrow = r + hi * 8;
            const size_t off  = (size_t)(b0 + mrow) * D_H + (h0 + n);
            const float  mv   = mem[r];
            __builtin_nontemporal_store((mv > THR) ? 1.0f : 0.0f, &sp[off]);
            __builtin_nontemporal_store(mv, &mp[off]);
        }
    }
}

// ---------------------------------------------------------------------------
extern "C" void kernel_launch(void* const* d_in, const int* in_sizes, int n_in,
                              void* d_out, int out_size, void* d_ws, size_t ws_size,
                              hipStream_t stream) {
    (void)in_sizes; (void)n_in; (void)out_size;
    const float* x = (const float*)d_in[0];  // [200,256,512]
    const float* W = (const float*)d_in[1];  // [512,512]
    const float* b = (const float*)d_in[2];  // [512]

    float* spk = (float*)d_out;
    float* mem = spk + NX;                   // T*B*D_H == NX here

    dim3 grid(D_H / 64, BATCH / 16);         // 8 x 16 = 128 blocks
    const size_t need_ws = (NX + NW) * sizeof(_Float16);

    if (ws_size >= need_ws) {
        _Float16* xh = (_Float16*)d_ws;
        _Float16* wh = xh + NX;
        const int ncvt4 = (int)((NX + NW) / 4);          // 6,619,136
        cvt_f16_kernel<<<(ncvt4 + 255) / 256, 256, 0, stream>>>(x, W, xh, wh);
        snn_fused_async_kernel<<<grid, 128, 0, stream>>>(xh, wh, b, spk, mem);
    } else {
        snn_fused_f32_kernel<<<grid, 128, 0, stream>>>(x, W, b, spk, mem);
    }
}
